// GroupDenseFull_22497038696904
// MI455X (gfx1250) — matrix operand order repaired and checked by reading, not compiled, not run on hardware
//
#include <hip/hip_runtime.h>
#include <hip/hip_bf16.h>
#include <stdint.h>

// GroupDenseFull: z[b,t,v] = sum_s kfull[s,t] * (sum_w x[b,s,w]*kseq[s,w,v])
// B=131072, C=1024, S=128, W=8. Memory-bound (~1 GB traffic, 46 us floor);
// stage-2 (34 GFLOP) on bf16 WMMA is far under that floor, so fuse everything.

#define B_TOT   131072
#define C_TOT   1024
#define W_SZ    8
#define S_SZ    128
#define ROWS    16        // batch rows per block
#define THREADS 256       // 8 wave32

typedef __attribute__((ext_vector_type(16))) __bf16 v16bf;
typedef __attribute__((ext_vector_type(8)))  float  v8f;

__device__ __forceinline__ uint16_t f32_bf16(float f) {
  union { float f; uint32_t u; } c; c.f = f;
  uint32_t u = c.u;
  u += 0x7FFFu + ((u >> 16) & 1u);      // round-to-nearest-even
  return (uint16_t)(u >> 16);
}

union Frag { v16bf v; uint32_t w[8]; };

__global__ __launch_bounds__(THREADS)
void GroupDenseFull_kernel(const float* __restrict__ x,
                           const float* __restrict__ kseq,
                           const float* __restrict__ kfull,
                           float* __restrict__ out) {
  // One flat 64 KB LDS block (safe static-LDS size), three overlapping views:
  //   [0      , 32 KB) : y  bf16 [v=8][b=16][s=128]  (XOR-swizzled on s-pairs)
  //   [32 KB  , 64 KB) : kf bf16 [s=128][t=128]      (XOR-swizzled on t-pairs)
  //   [0      , 64 KB) : z  f32  [b=16][c=1024]      (after WMMA, for coalesced store)
  __shared__ uint32_t lds[16384];
  uint16_t* y16  = (uint16_t*)lds;            // 16384 ushorts
  uint16_t* kf16 = ((uint16_t*)lds) + 16384;  // 16384 ushorts
  float*    zbuf = (float*)lds;

  const int tid  = threadIdx.x;
  const int b0   = blockIdx.x * ROWS;

  // Prefetch next block's x tile into L2 (speculative, gfx1250 global_prefetch_b8).
  if (blockIdx.x + 1 < gridDim.x) {
    const char* nx = (const char*)(x + (size_t)(b0 + ROWS) * C_TOT);
    __builtin_prefetch(nx + tid * 256, 0, 0);
  }

  // ---- Phase 1: kfull f32 -> bf16 LDS, layout kf[s][t] with XOR swizzle ----
  // phys ushort col for (s,t): (((t>>1) ^ s) & 63)*2 + (t&1)  -> dword col = (t>>1)^(s&63)
  for (int i = 0; i < (S_SZ * S_SZ) / THREADS; ++i) {
    int idx = i * THREADS + tid;
    int s = idx >> 7, t = idx & 127;
    int col = (((((t >> 1) ^ s) & 63) << 1) | (t & 1));
    kf16[s * 128 + col] = f32_bf16(kfull[idx]);
  }

  // ---- Phase 2: grouped 8x8 matmul (stage 1), y[b,s,v] -> bf16 LDS ----
  // Thread layout: s = tid & 127 fixed per thread -> kseq row loaded once.
  {
    const int s = tid & (S_SZ - 1);
    float kr[64];
    const float4* kq = (const float4*)(kseq + s * 64);
#pragma unroll
    for (int q = 0; q < 16; ++q) {
      float4 t4 = kq[q];
      kr[q * 4 + 0] = t4.x; kr[q * 4 + 1] = t4.y;
      kr[q * 4 + 2] = t4.z; kr[q * 4 + 3] = t4.w;
    }
#pragma unroll
    for (int i = 0; i < (ROWS * S_SZ) / THREADS; ++i) {   // 8 tasks/thread
      int task = i * THREADS + tid;
      int b = task >> 7;                                  // 0..15 (s == task&127)
      const float4* xp = (const float4*)(x + (size_t)(b0 + b) * C_TOT + s * W_SZ);
      float4 xa = xp[0], xb = xp[1];
      float x8[8] = {xa.x, xa.y, xa.z, xa.w, xb.x, xb.y, xb.z, xb.w};
      float yv[8];
#pragma unroll
      for (int v = 0; v < 8; ++v) yv[v] = 0.f;
#pragma unroll
      for (int w = 0; w < 8; ++w)
#pragma unroll
        for (int v = 0; v < 8; ++v)
          yv[v] = __builtin_fmaf(x8[w], kr[w * 8 + v], yv[v]);
      // y phys ushort col for (b,s): (((s>>1) ^ b) & 63)*2 + (s&1)
      int col = (((((s >> 1) ^ b) & 63) << 1) | (s & 1));
#pragma unroll
      for (int v = 0; v < 8; ++v)
        y16[v * 2048 + b * 128 + col] = f32_bf16(yv[v]);
    }
  }
  __syncthreads();

  // ---- Phase 3: stage 2 via v_wmma_f32_16x16x32_bf16 ----
  // Wave w owns v=w. Z_v(16x128) = Y_v(16x128) * Kf(128x128); 4 K-steps x 8 N-tiles.
  const int wave = tid >> 5;           // = v
  const int lane = tid & 31;
  const int m    = lane & 15;
  const int half = lane >> 4;

  v8f acc[8];
#pragma unroll
  for (int n = 0; n < 8; ++n) acc[n] = (v8f){0.f,0.f,0.f,0.f,0.f,0.f,0.f,0.f};

  const uint32_t* yd = lds;            // dword view: y  at [v*1024 + m*64 + q]
  const uint32_t* kd = lds + 8192;     // dword view: kf at [s*64 + q]

#pragma unroll
  for (int kk = 0; kk < 4; ++kk) {
    const int kbase = kk * 32;
    // A fragment (bf16 16x32): lanes 0-15 M=lane, halves split K per ISA table.
    Frag a;
#pragma unroll
    for (int r = 0; r < 8; ++r) {
      int kp = (r < 4) ? (half * 8 + 2 * r) : (16 + half * 8 + 2 * (r - 4));
      int qlog = (kbase + kp) >> 1;                 // logical dword col 0..63
      a.w[r] = yd[wave * 1024 + m * 64 + (qlog ^ m)];  // de-swizzle
    }
#pragma unroll
    for (int n = 0; n < 8; ++n) {
      Frag bfr;
      const int t = n * 16 + m;
#pragma unroll
      for (int r = 0; r < 8; ++r) {
        int kp = (r < 4) ? (half * 8 + 2 * r) : (16 + half * 8 + 2 * (r - 4));
        int k0 = kbase + kp, k1 = k0 + 1;
        int c0 = (((((t >> 1) ^ k0) & 63) << 1) | (t & 1));
        int c1 = (((((t >> 1) ^ k1) & 63) << 1) | (t & 1));
        bfr.w[r] = (uint32_t)kf16[k0 * 128 + c0] | ((uint32_t)kf16[k1 * 128 + c1] << 16);
      }
      acc[n] = __builtin_amdgcn_wmma_f32_16x16x32_bf16(
          false, a.v, false, bfr.v, (short)0, acc[n], false, false);
    }
  }
  __syncthreads();   // y/kf regions dead; reuse LDS as z staging

  // ---- Phase 4: C/D layout -> z LDS. (N=lane%16, M=vgpr + 8*(lane/16)) ----
#pragma unroll
  for (int n = 0; n < 8; ++n) {
    int t   = n * 16 + m;
    int col = t * W_SZ + wave;          // output channel = t*8 + v
#pragma unroll
    for (int r = 0; r < 8; ++r) {
      int Mrow = r + 8 * half;
      zbuf[Mrow * C_TOT + col] = acc[n][r];
    }
  }
  __syncthreads();

  // ---- Phase 5: fully coalesced float4 stores of the 16x1024 tile ----
  const float4* z4 = (const float4*)zbuf;
  float4* o4 = (float4*)(out + (size_t)b0 * C_TOT);
#pragma unroll
  for (int i = 0; i < (ROWS * C_TOT / 4) / THREADS; ++i) {  // 16 iters
    int idx = i * THREADS + tid;
    o4[idx] = z4[idx];
  }
}

extern "C" void kernel_launch(void* const* d_in, const int* in_sizes, int n_in,
                              void* d_out, int out_size, void* d_ws, size_t ws_size,
                              hipStream_t stream) {
  const float* x     = (const float*)d_in[0];   // (B, C) f32
  const float* kseq  = (const float*)d_in[1];   // (S, W, W) f32
  const float* kfull = (const float*)d_in[2];   // (S, S) f32
  float* out = (float*)d_out;                   // (B, C) f32
  dim3 grid(B_TOT / ROWS), block(THREADS);
  GroupDenseFull_kernel<<<grid, block, 0, stream>>>(x, kseq, kfull, out);
}
